// MLASelfAttention_36043365548097
// MI455X (gfx1250) — matrix operand order repaired and checked by reading, not compiled, not run on hardware
//
#include <hip/hip_runtime.h>
#include <cstdint>
#include <cstddef>

typedef __attribute__((ext_vector_type(16))) _Float16 v16h;
typedef __attribute__((ext_vector_type(8)))  _Float16 v8h;
typedef __attribute__((ext_vector_type(8)))  float    v8f;
typedef __attribute__((ext_vector_type(4)))  int      v4i;

// ---- optional CDNA5 async global->LDS path (compile-safe via __has_builtin) ----
#if defined(__has_builtin)
#if __has_builtin(__builtin_amdgcn_global_load_async_to_lds_b128) && \
    __has_builtin(__builtin_amdgcn_s_wait_asynccnt)
#define USE_ASYNC_COPY 1
#endif
#endif
#ifndef USE_ASYNC_COPY
#define USE_ASYNC_COPY 0
#endif

namespace {
constexpr int Bc   = 2;
constexpr int Lc   = 2048;
constexpr int Dc   = 2048;
constexpr int Hc   = 16;
constexpr int Kc   = 128;
constexpr int CKVc = 512;
constexpr int CQc  = 384;
constexpr int Rc   = 64;
constexpr int Mrows = Bc * Lc;            // 4096
constexpr float kScale = 0.07216878364870323f;  // 1/sqrt(K+R)=1/sqrt(192)
}

// -------------------------------------------------------------------------
// fragment loaders (WMMA 16x16x32 f16 layouts, wave32)
// A (16x32): lane m=lane%16 holds K = kb+{0..7} and kb+16+{0..7}, kb = lane<16?0:8
// B (32x16): lane n=lane%16 holds K = kb2+{0..7} and kb2+16+{0..7}, kb2 = lane<16?0:8
// C (16x16): VGPR v -> row v + (lane>=16 ? 8:0), col = lane%16
// -------------------------------------------------------------------------
__device__ inline v16h frag_a16(const _Float16* __restrict__ p) {
  const v8h lo = *(const v8h*)(p);
  const v8h hi = *(const v8h*)(p + 16);
  v16h r;
#pragma unroll
  for (int i = 0; i < 8; ++i) { r[i] = lo[i]; r[8 + i] = hi[i]; }
  return r;
}

__device__ inline v16h frag_b16(const _Float16* __restrict__ p) {
  const v8h lo = *(const v8h*)(p);
  const v8h hi = *(const v8h*)(p + 16);
  v16h r;
#pragma unroll
  for (int i = 0; i < 8; ++i) { r[i] = lo[i]; r[8 + i] = hi[i]; }
  return r;
}

__device__ inline v8f wmma_f16(v16h a, v16h b, v8f c) {
  return __builtin_amdgcn_wmma_f32_16x16x32_f16(false, a, false, b, (short)0, c,
                                                false, false);
}

// -------------------------------------------------------------------------
// elementwise f32 -> f16
// -------------------------------------------------------------------------
__global__ void cvt_f16_kernel(const float* __restrict__ src,
                               _Float16* __restrict__ dst, size_t n) {
  size_t i = (size_t)blockIdx.x * blockDim.x + threadIdx.x;
  if (i < n) dst[i] = (_Float16)src[i];
}

// W[rows][cols] f32 -> Wt[cols][rows] f16   (idx = c*rows + r, coalesced writes)
__global__ void wt_kernel(const float* __restrict__ src,
                          _Float16* __restrict__ dst, int rows, int cols) {
  size_t i = (size_t)blockIdx.x * blockDim.x + threadIdx.x;
  size_t n = (size_t)rows * cols;
  if (i >= n) return;
  int r = (int)(i % rows);
  int c = (int)(i / rows);
  dst[i] = (_Float16)src[(size_t)r * cols + c];
}

// -------------------------------------------------------------------------
// NT GEMM: C[M][N] = A[M][Kd] * Bt[N][Kd]^T   (A,Bt f16 row-major)
// block = 128 threads = 4 waves; tile 64(M) x 64(N); K step 32
// MODE 0: f16 row-major   MODE 1: f32 row-major   MODE 2: f16 transposed [N][M]
// -------------------------------------------------------------------------
template <int MODE>
__global__ void __launch_bounds__(128)
gemm_nt_kernel(const _Float16* __restrict__ A, const _Float16* __restrict__ Bt,
               void* __restrict__ Cout, int M, int N, int Kd) {
  __shared__ __align__(16) _Float16 sB[64 * 32];  // [n_local][k_local]
  const int wave = threadIdx.x >> 5;
  const int lane = threadIdx.x & 31;
  const int m0 = blockIdx.x * 64 + wave * 16;
  const int n0 = blockIdx.y * 64;
  const int nl = lane & 15;
  const int kb  = (lane < 16) ? 0 : 8;
  const int kb2 = (lane < 16) ? 0 : 8;

  v8f acc[4] = {};
  const _Float16* arow = A + (size_t)(m0 + nl) * Kd;

  for (int k0 = 0; k0 < Kd; k0 += 32) {
#if USE_ASYNC_COPY
    {
      const int t = threadIdx.x;
#pragma unroll
      for (int c = 0; c < 2; ++c) {
        const int ch  = t + c * 128;      // 256 chunks of 8 halfs
        const int row = ch >> 2;
        const int cho = (ch & 3) * 8;
        _Float16* src = const_cast<_Float16*>(Bt + (size_t)(n0 + row) * Kd + k0 + cho);
        __builtin_amdgcn_global_load_async_to_lds_b128(
            (__attribute__((address_space(1))) v4i*)src,
            (__attribute__((address_space(3))) v4i*)&sB[row * 32 + cho], 0, 0);
      }
      __builtin_amdgcn_s_wait_asynccnt(0);
    }
#else
    {
      const int t = threadIdx.x;
      const int row = t >> 1;
      const int ch  = (t & 1) * 16;
      const v8h* src = (const v8h*)(Bt + (size_t)(n0 + row) * Kd + k0 + ch);
      v8h* dst = (v8h*)(sB + row * 32 + ch);
      dst[0] = src[0];
      dst[1] = src[1];
    }
#endif
    __syncthreads();
    if (k0 + 32 < Kd) __builtin_prefetch(arow + k0 + 32, 0, 1);

    const v16h af = frag_a16(arow + k0 + kb);
#pragma unroll
    for (int t = 0; t < 4; ++t) {
      const v16h bf = frag_b16(sB + (t * 16 + nl) * 32 + kb2);
      acc[t] = wmma_f16(af, bf, acc[t]);
    }
    __syncthreads();
  }

  const int rhalf = (lane >> 4) << 3;
#pragma unroll
  for (int t = 0; t < 4; ++t) {
#pragma unroll
    for (int v = 0; v < 8; ++v) {
      const int gm = m0 + rhalf + v;
      const int gn = n0 + t * 16 + nl;
      if constexpr (MODE == 0)
        ((_Float16*)Cout)[(size_t)gm * N + gn] = (_Float16)acc[t][v];
      else if constexpr (MODE == 1)
        ((float*)Cout)[(size_t)gm * N + gn] = acc[t][v];
      else
        ((_Float16*)Cout)[(size_t)gn * M + gm] = (_Float16)acc[t][v];
    }
  }
}

// -------------------------------------------------------------------------
// in-place RoPE on f16 [rows][nh][64]; pos = row % L, half = 32
// -------------------------------------------------------------------------
__global__ void rope_kernel(_Float16* __restrict__ x, int rows, int nh) {
  size_t idx = (size_t)blockIdx.x * blockDim.x + threadIdx.x;
  size_t total = (size_t)rows * nh * 32;
  if (idx >= total) return;
  const int i = (int)(idx & 31);
  const size_t rh = idx >> 5;                 // row*nh + h
  const int pos = (int)((rh / nh) % Lc);
  const float inv = __powf(10000.0f, -(float)i / 32.0f);
  float sv, cv;
  __sincosf((float)pos * inv, &sv, &cv);
  _Float16* p = x + rh * 64;
  const float x1 = (float)p[i];
  const float x2 = (float)p[i + 32];
  p[i]      = (_Float16)(x1 * cv - x2 * sv);
  p[i + 32] = (_Float16)(x1 * sv + x2 * cv);
}

// -------------------------------------------------------------------------
// Flash-style causal attention. grid=(B*H, L/64), block=128 (4 waves).
// Each wave owns one 16-row tile; iterates 32-column steps with online softmax.
// Interior steps (strictly below the diagonal) skip causal compares entirely;
// exactly one final step per row-tile runs the masked path.
// Q  : [B*L][H*128] f16      Qr : [B*L][H*64] f16
// Kh : [B*L][H*128] f16      Kr : [B*L][64]   f16
// Vt : [H*128][B*L] f16 (transposed V)
// Out: [B*L][H*128] f16
// -------------------------------------------------------------------------
__global__ void __launch_bounds__(128)
mla_attn_kernel(const _Float16* __restrict__ Q, const _Float16* __restrict__ Qr,
                const _Float16* __restrict__ Kh, const _Float16* __restrict__ Kr,
                const _Float16* __restrict__ Vt,
                const unsigned char* __restrict__ kpm,
                _Float16* __restrict__ Out, float scale) {
  __shared__ __align__(16) _Float16 sP[4][16][32];

  const int bh = blockIdx.x;
  const int b = bh >> 4;
  const int h = bh & 15;
  const int wave = threadIdx.x >> 5;
  const int lane = threadIdx.x & 31;
  const int l0 = (blockIdx.y * 4 + wave) * 16;
  const int nl = lane & 15;
  const int rhalf = (lane >> 4) << 3;       // 0 or 8 (C-frag row offset)
  const int kb  = (lane < 16) ? 0 : 8;      // A-frag K base
  const int kb2 = (lane < 16) ? 0 : 8;      // B-frag K base

  // Q fragments for this 16-row tile (row = nl)
  const size_t qrow  = (size_t)(b * Lc + l0 + nl) * (Hc * Kc) + h * Kc;
  const size_t qrrow = (size_t)(b * Lc + l0 + nl) * (Hc * Rc) + h * Rc;
  v16h qf[4], qrf[2];
#pragma unroll
  for (int t = 0; t < 4; ++t) qf[t] = frag_a16(Q + qrow + t * 32 + kb);
#pragma unroll
  for (int t = 0; t < 2; ++t) qrf[t] = frag_a16(Qr + qrrow + t * 32 + kb);

  v8f o[8] = {};
  float mrow[8], lrow[8];
#pragma unroll
  for (int v = 0; v < 8; ++v) { mrow[v] = -1e30f; lrow[v] = 0.0f; }

  auto step = [&](int c0, bool diag) {
    v8f s0 = {}, s1 = {};

    // S = q . k^T   (columns c0+nl and c0+16+nl)
    const size_t ka  = (size_t)(b * Lc + c0 + nl) * (Hc * Kc) + h * Kc + kb2;
    const size_t kbp = (size_t)(b * Lc + c0 + 16 + nl) * (Hc * Kc) + h * Kc + kb2;
#pragma unroll
    for (int t = 0; t < 4; ++t) {
      s0 = wmma_f16(qf[t], frag_b16(Kh + ka + t * 32), s0);
      s1 = wmma_f16(qf[t], frag_b16(Kh + kbp + t * 32), s1);
    }
    const size_t ra = (size_t)(b * Lc + c0 + nl) * Rc + kb2;
    const size_t rb = (size_t)(b * Lc + c0 + 16 + nl) * Rc + kb2;
#pragma unroll
    for (int t = 0; t < 2; ++t) {
      s0 = wmma_f16(qrf[t], frag_b16(Kr + ra + t * 32), s0);
      s1 = wmma_f16(qrf[t], frag_b16(Kr + rb + t * 32), s1);
    }

    // key-padding mask folded into an additive bias (kpm is per-column)
    const float bias0 = (kpm[b * Lc + c0 + nl] != 0) ? -1e30f : 0.0f;
    const float bias1 = (kpm[b * Lc + c0 + 16 + nl] != 0) ? -1e30f : 0.0f;
    const int gc0 = c0 + nl;
    const int gc1 = c0 + 16 + nl;

    float alph[8];
#pragma unroll
    for (int v = 0; v < 8; ++v) {
      float x0 = s0[v] * scale + bias0;
      float x1 = s1[v] * scale + bias1;
      if (diag) {
        const int gr = l0 + rhalf + v;
        if (gc0 > gr) x0 = -1e30f;
        if (gc1 > gr) x1 = -1e30f;
      }

      float t = fmaxf(x0, x1);
      t = fmaxf(t, __shfl_xor(t, 1, 16));
      t = fmaxf(t, __shfl_xor(t, 2, 16));
      t = fmaxf(t, __shfl_xor(t, 4, 16));
      t = fmaxf(t, __shfl_xor(t, 8, 16));
      const float mnew = fmaxf(mrow[v], t);
      alph[v] = __expf(mrow[v] - mnew);
      const float p0 = __expf(x0 - mnew);
      const float p1 = __expf(x1 - mnew);
      float rs = p0 + p1;
      rs += __shfl_xor(rs, 1, 16);
      rs += __shfl_xor(rs, 2, 16);
      rs += __shfl_xor(rs, 4, 16);
      rs += __shfl_xor(rs, 8, 16);
      lrow[v] = lrow[v] * alph[v] + rs;
      mrow[v] = mnew;
      sP[wave][rhalf + v][nl]      = (_Float16)p0;
      sP[wave][rhalf + v][16 + nl] = (_Float16)p1;
    }

    // rescale O accumulators
#pragma unroll
    for (int t = 0; t < 8; ++t)
#pragma unroll
      for (int v = 0; v < 8; ++v) o[t][v] *= alph[v];

    // P fragment from LDS (intra-wave, in-order LDS => no block barrier)
    const v16h pf = frag_a16(&sP[wave][nl][kb]);

    // O += P . V
#pragma unroll
    for (int t = 0; t < 8; ++t) {
      const _Float16* vp =
          Vt + (size_t)(h * Kc + t * 16 + nl) * Mrows + (size_t)b * Lc + c0 + kb2;
      o[t] = wmma_f16(pf, frag_b16(vp), o[t]);
    }
  };

  // interior steps: every column strictly <= every row, no causal compares
  const int nfull = l0 >> 5;
  for (int st = 0; st < nfull; ++st) step(st * 32, false);
  // single diagonal step (always exists: covers columns [32*nfull, 32*nfull+31])
  step(nfull * 32, true);

  // normalize and store
#pragma unroll
  for (int t = 0; t < 8; ++t)
#pragma unroll
    for (int v = 0; v < 8; ++v) {
      const float l = lrow[v];
      const float val = (l > 0.0f) ? (o[t][v] / l) : 0.0f;
      Out[(size_t)(b * Lc + l0 + rhalf + v) * (Hc * Kc) + h * Kc + t * 16 + nl] =
          (_Float16)val;
    }
}

// -------------------------------------------------------------------------
// host launcher
// -------------------------------------------------------------------------
extern "C" void kernel_launch(void* const* d_in, const int* in_sizes, int n_in,
                              void* d_out, int out_size, void* d_ws, size_t ws_size,
                              hipStream_t stream) {
  const float* x = (const float*)d_in[0];
  const unsigned char* kpm = (const unsigned char*)d_in[1];
  const float* W_kvc = (const float*)d_in[2];
  const float* W_dq  = (const float*)d_in[3];
  const float* W_uq  = (const float*)d_in[4];
  const float* W_qr  = (const float*)d_in[5];
  const float* W_kr  = (const float*)d_in[6];
  const float* W_uk  = (const float*)d_in[7];
  const float* W_uv  = (const float*)d_in[8];
  const float* W_o   = (const float*)d_in[9];
  float* out = (float*)d_out;

  _Float16* p = (_Float16*)d_ws;
  auto alloc = [&](size_t n) { _Float16* r = p; p += n; return r; };
  _Float16* xh     = alloc((size_t)Mrows * Dc);
  _Float16* Wt_kvc = alloc((size_t)Dc * CKVc);
  _Float16* Wt_dq  = alloc((size_t)Dc * CQc);
  _Float16* Wt_uq  = alloc((size_t)CQc * Hc * Kc);
  _Float16* Wt_qr  = alloc((size_t)CQc * Hc * Rc);
  _Float16* Wt_kr  = alloc((size_t)Dc * Rc);
  _Float16* Wt_uk  = alloc((size_t)CKVc * Hc * Kc);
  _Float16* Wt_uv  = alloc((size_t)CKVc * Hc * Kc);
  _Float16* Wt_o   = alloc((size_t)Hc * Kc * Dc);
  _Float16* kv_c   = alloc((size_t)Mrows * CKVc);
  _Float16* q_c    = alloc((size_t)Mrows * CQc);
  _Float16* qbuf   = alloc((size_t)Mrows * Hc * Kc);
  _Float16* q_r    = alloc((size_t)Mrows * Hc * Rc);
  _Float16* kh     = alloc((size_t)Mrows * Hc * Kc);
  _Float16* vt     = alloc((size_t)Hc * Kc * Mrows);
  _Float16* k_r    = alloc((size_t)Mrows * Rc);
  _Float16* attno  = alloc((size_t)Mrows * Hc * Kc);

  const int TB = 256;
  auto gb = [&](size_t n) { return dim3((unsigned)((n + TB - 1) / TB)); };

  // stage 0: conversions / weight transposes
  cvt_f16_kernel<<<gb((size_t)Mrows * Dc), TB, 0, stream>>>(x, xh, (size_t)Mrows * Dc);
  wt_kernel<<<gb((size_t)Dc * CKVc), TB, 0, stream>>>(W_kvc, Wt_kvc, Dc, CKVc);
  wt_kernel<<<gb((size_t)Dc * CQc), TB, 0, stream>>>(W_dq, Wt_dq, Dc, CQc);
  wt_kernel<<<gb((size_t)CQc * Hc * Kc), TB, 0, stream>>>(W_uq, Wt_uq, CQc, Hc * Kc);
  wt_kernel<<<gb((size_t)CQc * Hc * Rc), TB, 0, stream>>>(W_qr, Wt_qr, CQc, Hc * Rc);
  wt_kernel<<<gb((size_t)Dc * Rc), TB, 0, stream>>>(W_kr, Wt_kr, Dc, Rc);
  wt_kernel<<<gb((size_t)CKVc * Hc * Kc), TB, 0, stream>>>(W_uk, Wt_uk, CKVc, Hc * Kc);
  wt_kernel<<<gb((size_t)CKVc * Hc * Kc), TB, 0, stream>>>(W_uv, Wt_uv, CKVc, Hc * Kc);
  wt_kernel<<<gb((size_t)Hc * Kc * Dc), TB, 0, stream>>>(W_o, Wt_o, Hc * Kc, Dc);

  const dim3 blk(128);
  // stage 1: projections from x
  gemm_nt_kernel<0><<<dim3(Mrows / 64, CKVc / 64), blk, 0, stream>>>(xh, Wt_kvc, kv_c, Mrows, CKVc, Dc);
  gemm_nt_kernel<0><<<dim3(Mrows / 64, CQc / 64), blk, 0, stream>>>(xh, Wt_dq, q_c, Mrows, CQc, Dc);
  gemm_nt_kernel<0><<<dim3(Mrows / 64, Rc / 64), blk, 0, stream>>>(xh, Wt_kr, k_r, Mrows, Rc, Dc);
  // stage 2: up-projections
  gemm_nt_kernel<0><<<dim3(Mrows / 64, (Hc * Kc) / 64), blk, 0, stream>>>(q_c, Wt_uq, qbuf, Mrows, Hc * Kc, CQc);
  gemm_nt_kernel<0><<<dim3(Mrows / 64, (Hc * Rc) / 64), blk, 0, stream>>>(q_c, Wt_qr, q_r, Mrows, Hc * Rc, CQc);
  gemm_nt_kernel<0><<<dim3(Mrows / 64, (Hc * Kc) / 64), blk, 0, stream>>>(kv_c, Wt_uk, kh, Mrows, Hc * Kc, CKVc);
  gemm_nt_kernel<2><<<dim3(Mrows / 64, (Hc * Kc) / 64), blk, 0, stream>>>(kv_c, Wt_uv, vt, Mrows, Hc * Kc, CKVc);
  // rope (in place)
  rope_kernel<<<gb((size_t)Mrows * Hc * 32), TB, 0, stream>>>(q_r, Mrows, Hc);
  rope_kernel<<<gb((size_t)Mrows * 32), TB, 0, stream>>>(k_r, Mrows, 1);
  // stage 3: attention
  mla_attn_kernel<<<dim3(Bc * Hc, Lc / 64), blk, 0, stream>>>(qbuf, q_r, kh, k_r, vt, kpm, attno, kScale);
  // stage 4: output projection (f32 out)
  gemm_nt_kernel<1><<<dim3(Mrows / 64, Dc / 64), blk, 0, stream>>>(attno, Wt_o, out, Mrows, Dc, Hc * Kc);
}
